// HomoMaskSAM_88467736363590
// MI455X (gfx1250) — 8-bit matrix operand order repaired and checked by reading, compiled, not run on hardware
//
#include <hip/hip_runtime.h>
#include <hip/hip_bf16.h>


#define BB 8
#define NN 32
#define HH 512
#define WW 512
#define PP (HH * WW)
#define PSHIFT 18

typedef int v8i __attribute__((ext_vector_type(8)));

__device__ __forceinline__ float triw(int i) {
    const float w[9] = {1.f, 2.f, 3.f, 4.f, 5.f, 4.f, 3.f, 2.f, 1.f};
    return w[i];
}

__global__ __launch_bounds__(256) void k_rowblur(const float* __restrict__ x,
                                                 float* __restrict__ t1) {
    const int idx = blockIdx.x * 256 + threadIdx.x;
    const int xx = idx & (WW - 1);
    float s = 0.f;
#pragma unroll
    for (int d = -4; d <= 4; ++d) {
        const int xc = xx + d;
        if (xc >= 0 && xc < WW) s += triw(d + 4) * x[idx + d];
    }
    t1[idx] = s;
}

__global__ __launch_bounds__(256) void k_colblur_absdiff(const float* __restrict__ x,
                                                         const float* __restrict__ t1,
                                                         float* __restrict__ m) {
    const int idx = blockIdx.x * 256 + threadIdx.x;
    const int y = (idx >> 9) & (HH - 1);
    float s = 0.f;
#pragma unroll
    for (int d = -4; d <= 4; ++d) {
        const int yc = y + d;
        if (yc >= 0 && yc < HH) s += triw(d + 4) * t1[idx + d * WW];
    }
    const float mean = s * (1.f / 625.f);
    m[idx] = fabsf(x[idx] - mean);
}

__global__ __launch_bounds__(256) void k_rowmax(const float* __restrict__ m,
                                                float* __restrict__ t2) {
    const int idx = blockIdx.x * 256 + threadIdx.x;
    const int xx = idx & (WW - 1);
    float v = -3.4e38f;
#pragma unroll
    for (int d = -2; d <= 2; ++d) {
        const int xc = xx + d;
        if (xc >= 0 && xc < WW) v = fmaxf(v, m[idx + d]);
    }
    t2[idx] = v;
}

__global__ __launch_bounds__(256) void k_colmax_thresh(const float* __restrict__ t2,
                                                       unsigned char* __restrict__ notref) {
    const int idx = blockIdx.x * 256 + threadIdx.x;
    const int y = (idx >> 9) & (HH - 1);
    float v = -3.4e38f;
#pragma unroll
    for (int d = -2; d <= 2; ++d) {
        const int yc = y + d;
        if (yc >= 0 && yc < HH) v = fmaxf(v, t2[idx + d * WW]);
    }
    notref[idx] = (v >= 0.25f) ? (unsigned char)0 : (unsigned char)1;
}

__global__ void k_init(int* __restrict__ interBuf, int* __restrict__ totBuf,
                       unsigned* __restrict__ selbits) {
    const int t = threadIdx.x;
    if (t < 256) interBuf[t] = 0;
    else         totBuf[t - 256] = 0;
    if (t < BB)  selbits[t] = 0;
}

__device__ __forceinline__ unsigned msb4(const float4 f) {
    const unsigned f0 = __float_as_uint(f.x), f1 = __float_as_uint(f.y);
    const unsigned f2 = __float_as_uint(f.z), f3 = __float_as_uint(f.w);
    const unsigned t01 = __builtin_amdgcn_perm(f1, f0, 0x0C0C0703u);
    const unsigned t23 = __builtin_amdgcn_perm(f3, f2, 0x0C0C0703u);
    return __builtin_amdgcn_perm(t23, t01, 0x05040100u);
}

__device__ __forceinline__ v8i pack_A(const float* __restrict__ rowp, int half) {
    v8i Av;
#pragma unroll
    for (int j = 0; j < 8; ++j) {
        const int kst = ((j >> 2) << 5) | (half << 4) | ((j & 3) << 2);
        Av[j] = (int)msb4(*reinterpret_cast<const float4*>(rowp + kst));
    }
    return Av;
}

__device__ __forceinline__ v8i pack_B(const unsigned char* __restrict__ nrp, int half,
                                      unsigned m0, unsigned cbase) {
    const uint4 lo = *reinterpret_cast<const uint4*>(nrp + half * 16);
    const uint4 hi = *reinterpret_cast<const uint4*>(nrp + 32 + half * 16);
    v8i Bv;
    Bv[0] = (int)((lo.x & m0) | cbase);
    Bv[1] = (int)((lo.y & m0) | cbase);
    Bv[2] = (int)((lo.z & m0) | cbase);
    Bv[3] = (int)((lo.w & m0) | cbase);
    Bv[4] = (int)((hi.x & m0) | cbase);
    Bv[5] = (int)((hi.y & m0) | cbase);
    Bv[6] = (int)((hi.z & m0) | cbase);
    Bv[7] = (int)((hi.w & m0) | cbase);
    return Bv;
}

#define CNT_ITERS 8
__global__ __launch_bounds__(256) void k_counts(const float* __restrict__ sam,
                                                const unsigned char* __restrict__ notref,
                                                int* __restrict__ interBuf,
                                                int* __restrict__ totBuf) {
    const int wave  = threadIdx.x >> 5;
    const int lane  = threadIdx.x & 31;
    const int half  = lane >> 4;
    const int mrow  = lane & 15;

    const unsigned m0    = (mrow == 0) ? 0xFFFFFFFFu : 0u;
    const unsigned cbase = (mrow == 1) ? 0x01010101u : 0u;

    const int chunk = blockIdx.x & 63;
    const int g     = (blockIdx.x >> 6) & 1;
    const int b     = blockIdx.x >> 7;
    const int n0    = g * 16;
    const int n     = n0 + mrow;

    const float* rowbase = sam + (((size_t)(b * NN + n)) << PSHIFT);
    const unsigned char* nrbase = notref + ((size_t)b << PSHIFT);

    const int pbase = chunk * (CNT_ITERS * 64 * 8) + wave * (CNT_ITERS * 64);

    v8i acc = {0, 0, 0, 0, 0, 0, 0, 0};

#pragma unroll 1
    for (int it = 0; it < CNT_ITERS; ++it) {
        const int p = pbase + it * 64;
        v8i Av = pack_A(rowbase + p, half);
        v8i Bv = pack_B(nrbase + p, half, m0, cbase);
        acc = __builtin_amdgcn_wmma_i32_16x16x64_iu8(false, Av, false, Bv, acc,
                                                     false, false);
    }

    if (mrow < 2) {
        int* dst = (mrow == 0) ? interBuf : totBuf;
        const int nb = b * NN + n0 + (half ? 8 : 0);
#pragma unroll
        for (int r = 0; r < 8; ++r) atomicAdd(&dst[nb + r], acc[r]);
    }
}

__global__ void k_select(const int* __restrict__ interBuf,
                         const int* __restrict__ totBuf,
                         unsigned* __restrict__ selbits) {
    const int t = threadIdx.x;
    const int b = t >> 5;
    const int nn = t & 31;
    const int inter = interBuf[t];
    int tot = totBuf[t];
    if (tot < 63) tot = 63;
    if (5 * inter >= 4 * tot) atomicOr(&selbits[b], 1u << nn);
}

__global__ __launch_bounds__(256) void k_merge(const float* __restrict__ sam,
                                               const unsigned* __restrict__ selbits,
                                               float* __restrict__ out) {
    const int idx4 = blockIdx.x * 256 + threadIdx.x;
    const int b = idx4 >> (PSHIFT - 2);
    const int p = (idx4 & ((PP >> 2) - 1)) << 2;
    const unsigned sel = selbits[b];

    float4 o = make_float4(1.f, 1.f, 1.f, 1.f);
#pragma unroll 1
    for (int n = 0; n < NN; ++n) {
        if ((sel >> n) & 1u) {
            const float4 s = *reinterpret_cast<const float4*>(
                sam + (((size_t)(b * NN + n)) << PSHIFT) + p);
            if (s.x > 0.f) o.x = 0.f;
            if (s.y > 0.f) o.y = 0.f;
            if (s.z > 0.f) o.z = 0.f;
            if (s.w > 0.f) o.w = 0.f;
        }
    }
    *reinterpret_cast<float4*>(out + ((size_t)b << PSHIFT) + p) = o;
}

extern "C" void kernel_launch(void* const* d_in, const int* in_sizes, int n_in,
                              void* d_out, int out_size, void* d_ws, size_t ws_size,
                              hipStream_t stream) {
    const float* x   = (const float*)d_in[0];
    const float* sam = (const float*)d_in[1];
    float* out = (float*)d_out;

    char* ws = (char*)d_ws;
    float*         bufA    = (float*)(ws);
    float*         bufB    = (float*)(ws + (8u << 20));
    unsigned char* notref  = (unsigned char*)(ws + (8u << 20));
    int*           interB  = (int*)(ws + (10u << 20));
    int*           totB    = (int*)(ws + (10u << 20) + 1024);
    unsigned*      selbits = (unsigned*)(ws + (10u << 20) + 2048);

    const int nPix     = BB * PP;
    const int gStencil = nPix / 256;

    k_init<<<1, 512, 0, stream>>>(interB, totB, selbits);

    k_rowblur<<<gStencil, 256, 0, stream>>>(x, bufA);
    k_colblur_absdiff<<<gStencil, 256, 0, stream>>>(x, bufA, bufB);
    k_rowmax<<<gStencil, 256, 0, stream>>>(bufB, bufA);
    k_colmax_thresh<<<gStencil, 256, 0, stream>>>(bufA, notref);

    k_counts<<<BB * 2 * 64, 256, 0, stream>>>(sam, notref, interB, totB);

    k_select<<<1, 256, 0, stream>>>(interB, totB, selbits);

    k_merge<<<(BB * PP / 4) / 256, 256, 0, stream>>>(sam, selbits, out);
}
